// GeoAggregator_65962107731997
// MI455X (gfx1250) — matrix operand order repaired and checked by reading, not compiled, not run on hardware
//
#include <hip/hip_runtime.h>
#include <hip/hip_bf16.h>
#include <math.h>

typedef _Float16 half_t;
typedef __attribute__((ext_vector_type(16))) _Float16 v16h;
typedef __attribute__((ext_vector_type(8)))  _Float16 v8h;
typedef __attribute__((ext_vector_type(8)))  float    v8f;

#define BS    1024
#define SL    128
#define XD    8
#define DM    128
#define NH    8
#define DH    16
#define DHALF 64

// padded LDS strides (bank-conflict avoidance)
#define TSTR 136   // tokens / scores / v^T rows, halfs (272 B -> 4-bank step)
#define QSTR 40    // q_rot/k_rot & xpad rows, halfs (80 B -> 20-bank step)
#define HSTR 72    // hidden-layer rows, halfs (144 B -> 36-bank step)

// ---- flat LDS layout (bytes). Tokenizer scratch aliases attention bufs. ----
#define SM_TOK   0        // half [128][136]  34816 B (persistent tokens)
#define SM_SCB   34816    // half [128][136]  34816 B (scores/attn)
#define SM_QH    69632    // half [128][40]   10240 B (q_rot, K-pad 16..31 = 0)
#define SM_KH    79872    // half [128][40]   10240 B (k_rot, K-pad 16..31 = 0)
#define SM_VT    90112    // half [16][136]    4352 B (V transposed)
#define SM_SIN   94464    // float[128][8]     4096 B
#define SM_COS   98560    // float[128][8]     4096 B
#define SM_POOL  102656   // float[128]         512 B
#define SM_H1    103168   // float[64]
#define SM_H2    103424   // float[32]
#define SM_TOTAL 103552
// tokenizer-phase aliases (dead once the heads loop starts):
#define SM_HX    34816    // half [128][72]   18432 B (over scB)
#define SM_HY    53248    // half [128][72]   18432 B (over scB tail + qh16 head)
#define SM_XPAD  79872    // half [128][40]   10240 B (over kh16)

// ws layout (halfs)
#define WS_WQ   0
#define WS_WK   16384
#define WS_WV   32768
#define WS_W1X  49152   // [64][32], K-padded (rows of W1x at K=0..7)
#define WS_W1Y  51200   // [64][32], W1y at K=8
#define WS_W2X  53248   // [64][64]
#define WS_W2Y  57344   // [64][64]
#define WS_TOT  61440

__device__ inline v16h join16(v8h lo, v8h hi) {
  v16h r;
#pragma unroll
  for (int i = 0; i < 8; ++i) { r[i] = lo[i]; r[i + 8] = hi[i]; }
  return r;
}

// lane <-> lane^1 exchange via DPP quad_perm(1,0,3,2): stays on the VALU pipe
__device__ inline float swap_pair(float v) {
  int i = __builtin_bit_cast(int, v);
  int r = __builtin_amdgcn_update_dpp(0, i, 0xB1, 0xF, 0xF, true);
  return __builtin_bit_cast(float, r);
}

// Branchless tanh: native v_tanh_f32 on gfx1250 if the builtin exists,
// else exp-based identity (saturates correctly: exp->inf gives 1, exp->0 gives -1).
__device__ inline float fast_tanh(float x) {
#if __has_builtin(__builtin_amdgcn_tanhf)
  return __builtin_amdgcn_tanhf(x);
#else
  float t = __expf(2.0f * x);
  return 1.0f - 2.0f / (t + 1.0f);
#endif
}

__device__ inline float tanhshrink(float x) { return x - fast_tanh(x); }

// Pre-convert all GEMM weights to f16, transposed to [N][K] (and K-padded for
// layer 1) so every WMMA B-operand fetch is a contiguous 16-half run per lane.
__global__ void geo_wprep(const float* __restrict__ Wq, const float* __restrict__ Wk,
                          const float* __restrict__ Wv, const float* __restrict__ W1x,
                          const float* __restrict__ W1y, const float* __restrict__ W2x,
                          const float* __restrict__ W2y, half_t* __restrict__ wt) {
  int idx = blockIdx.x * blockDim.x + threadIdx.x;
  if (idx < WS_W1X) {                       // Wq/Wk/Wv : [128][128] -> [N][K]
    int m = idx / (DM * DM);
    int r = idx - m * (DM * DM);
    int n = r / DM, k = r - (r / DM) * DM;
    const float* W = (m == 0) ? Wq : ((m == 1) ? Wk : Wv);
    wt[idx] = (half_t)W[k * DM + n];
  } else if (idx < WS_W1Y) {                // W1x : [8][64] -> [64][32] K-pad
    int r = idx - WS_W1X;
    int n = r >> 5, k = r & 31;
    wt[idx] = (k < XD) ? (half_t)W1x[k * DHALF + n] : (half_t)0.0f;
  } else if (idx < WS_W2X) {                // W1y : [1][64] -> [64][32] at K=8
    int r = idx - WS_W1Y;
    int n = r >> 5, k = r & 31;
    wt[idx] = (k == XD) ? (half_t)W1y[n] : (half_t)0.0f;
  } else if (idx < WS_W2Y) {                // W2x : [64][64] -> [64][64]
    int r = idx - WS_W2X;
    int n = r >> 6, k = r & 63;
    wt[idx] = (half_t)W2x[k * DHALF + n];
  } else if (idx < WS_TOT) {                // W2y
    int r = idx - WS_W2Y;
    int n = r >> 6, k = r & 63;
    wt[idx] = (half_t)W2y[k * DHALF + n];
  }
}

__global__ __launch_bounds__(256, 1) void geo_main(
    const float* __restrict__ x, const float* __restrict__ y,
    const float* __restrict__ coords,
    const float* __restrict__ b1x, const float* __restrict__ b2x,
    const float* __restrict__ b1y, const float* __restrict__ b2y,
    const float* __restrict__ learnY,
    const half_t* __restrict__ WtQ, const half_t* __restrict__ WtK,
    const half_t* __restrict__ WtV,
    const half_t* __restrict__ Wt1x, const half_t* __restrict__ Wt1y,
    const half_t* __restrict__ Wt2x, const half_t* __restrict__ Wt2y,
    const float* __restrict__ bq, const float* __restrict__ bk,
    const float* __restrict__ bv,
    const float* __restrict__ dW1, const float* __restrict__ db1,
    const float* __restrict__ dW2, const float* __restrict__ db2,
    const float* __restrict__ dW3, const float* __restrict__ db3,
    float* __restrict__ out) {
  __shared__ __attribute__((aligned(128))) char smem[SM_TOTAL];
  half_t* tokS  = (half_t*)(smem + SM_TOK);
  half_t* scB   = (half_t*)(smem + SM_SCB);
  half_t* qh16  = (half_t*)(smem + SM_QH);
  half_t* kh16  = (half_t*)(smem + SM_KH);
  half_t* vT16  = (half_t*)(smem + SM_VT);
  float*  sinS  = (float*)(smem + SM_SIN);
  float*  cosS  = (float*)(smem + SM_COS);
  float*  poolS = (float*)(smem + SM_POOL);
  float*  h1S   = (float*)(smem + SM_H1);
  float*  h2S   = (float*)(smem + SM_H2);
  half_t* xpad  = (half_t*)(smem + SM_XPAD);
  half_t* hxS   = (half_t*)(smem + SM_HX);
  half_t* hyS   = (half_t*)(smem + SM_HY);

  const int b    = blockIdx.x;
  const int tid  = threadIdx.x;
  const int lane = tid & 31;
  const int wv   = tid >> 5;    // wave 0..7 -> M-tile row
  const int hf   = lane >> 4;   // half-wave
  const int lr   = lane & 15;
  const int arow = wv * 16 + lr;  // A-operand row owned by this lane

  // warm L2 with the transposed weights (global_prefetch_b8)
  __builtin_prefetch((const char*)WtQ + tid * 128, 0, 1);
  __builtin_prefetch((const char*)WtK + tid * 128, 0, 1);
  __builtin_prefetch((const char*)WtV + tid * 128, 0, 1);

  // ---- phase 1: stage z = [x | y | 0] as f16, rotary tables ----
  if (tid < SL) {
    const int t = tid;
    const float* xp = x + ((size_t)b * SL + t) * XD;
#pragma unroll
    for (int c = 0; c < XD; ++c) xpad[t * QSTR + c] = (half_t)xp[c];
    xpad[t * QSTR + XD] =
        (t < SL - 1) ? (half_t)y[(size_t)b * (SL - 1) + t] : (half_t)0.0f;
#pragma unroll
    for (int c = XD + 1; c < 32; ++c) xpad[t * QSTR + c] = (half_t)0.0f;
    { // 2D rotary tables: inv_freq = 10000^(-i/4) = {1, .1, .01, .001}
      const float invf[4] = {1.0f, 1.0e-1f, 1.0e-2f, 1.0e-3f};
      float cx = coords[((size_t)b * SL + t) * 2 + 0];
      float cy = coords[((size_t)b * SL + t) * 2 + 1];
#pragma unroll
      for (int i = 0; i < 4; ++i) {
        sinS[t * 8 + i]     = __sinf(cx * invf[i]);
        sinS[t * 8 + 4 + i] = __sinf(cy * invf[i]);
        cosS[t * 8 + i]     = __cosf(cx * invf[i]);
        cosS[t * 8 + 4 + i] = __cosf(cy * invf[i]);
      }
    }
    poolS[t] = 0.0f;
  }
  __syncthreads();

  // ---- phase 2: tokenizer layer 1 (two 128x64, K=32 GEMMs) + tanhshrink ----
  {
    const half_t* ap = &xpad[arow * QSTR + hf * 16];
    const v16h az = join16(*(const v8h*)ap, *(const v8h*)(ap + 8));
#pragma unroll
    for (int nt = 0; nt < 4; ++nt) {
      const int ncol = nt * 16 + lr;
      const half_t* bxp = Wt1x + ncol * 32 + hf * 16;
      const half_t* byp = Wt1y + ncol * 32 + hf * 16;
      v16h bx = join16(*(const v8h*)bxp, *(const v8h*)(bxp + 8));
      v16h by = join16(*(const v8h*)byp, *(const v8h*)(byp + 8));
      v8f cx1, cy1;
#pragma unroll
      for (int i = 0; i < 8; ++i) { cx1[i] = b1x[ncol]; cy1[i] = b1y[ncol]; }
      cx1 = __builtin_amdgcn_wmma_f32_16x16x32_f16(false, az, false, bx, (short)0, cx1, false, false);
      cy1 = __builtin_amdgcn_wmma_f32_16x16x32_f16(false, az, false, by, (short)0, cy1, false, false);
#pragma unroll
      for (int j = 0; j < 8; ++j) {
        const int m = wv * 16 + hf * 8 + j;
        hxS[m * HSTR + ncol] = (half_t)tanhshrink(cx1[j]);
        hyS[m * HSTR + ncol] = (half_t)tanhshrink(cy1[j]);
      }
    }
  }
  __syncthreads();

  // ---- phase 3: tokenizer layer 2 (two 128x64, K=64 GEMMs) -> tokens ----
#pragma unroll
  for (int nt = 0; nt < 4; ++nt) {
    const int ncol = nt * 16 + lr;
    v8f ctx, cty;
#pragma unroll
    for (int i = 0; i < 8; ++i) { ctx[i] = b2x[ncol]; cty[i] = b2y[ncol]; }
#pragma unroll
    for (int kk = 0; kk < 2; ++kk) {
      const int kb = kk * 32;
      const half_t* axp = &hxS[arow * HSTR + kb + hf * 16];
      const half_t* ayp = &hyS[arow * HSTR + kb + hf * 16];
      v16h ax = join16(*(const v8h*)axp, *(const v8h*)(axp + 8));
      v16h ay = join16(*(const v8h*)ayp, *(const v8h*)(ayp + 8));
      const half_t* bxp = Wt2x + ncol * 64 + kb + hf * 16;
      const half_t* byp = Wt2y + ncol * 64 + kb + hf * 16;
      v16h bx = join16(*(const v8h*)bxp, *(const v8h*)(bxp + 8));
      v16h by = join16(*(const v8h*)byp, *(const v8h*)(byp + 8));
      ctx = __builtin_amdgcn_wmma_f32_16x16x32_f16(false, ax, false, bx, (short)0, ctx, false, false);
      cty = __builtin_amdgcn_wmma_f32_16x16x32_f16(false, ay, false, by, (short)0, cty, false, false);
    }
#pragma unroll
    for (int j = 0; j < 8; ++j) {
      const int m = wv * 16 + hf * 8 + j;
      tokS[m * TSTR + ncol]         = (half_t)ctx[j];
      tokS[m * TSTR + DHALF + ncol] = (half_t)cty[j];
    }
  }
  __syncthreads();

  // ---- phase 4: learnable_y row + zero the q/k K-pads (aliased bufs now dead)
  if (tid < DHALF) tokS[(SL - 1) * TSTR + DHALF + tid] = (half_t)learnY[tid];
  if (tid < SL) {
#pragma unroll
    for (int c = DH; c < 2 * DH; ++c) {
      qh16[tid * QSTR + c] = (half_t)0.0f;
      kh16[tid * QSTR + c] = (half_t)0.0f;
    }
  }
  __syncthreads();

  // ---------------- attention, head by head ----------------
  for (int h = 0; h < NH; ++h) {
    // ---- QKV slice for this head: [128x128] @ [128x16] via f16 WMMA ----
    const int ncol = h * DH + lr;
    v8f cq, ck, cv;
#pragma unroll
    for (int i = 0; i < 8; ++i) { cq[i] = bq[ncol]; ck[i] = bk[ncol]; cv[i] = bv[ncol]; }
#pragma unroll
    for (int kk = 0; kk < 4; ++kk) {
      const int kb = kk * 32;
      const half_t* ap = &tokS[arow * TSTR + kb + hf * 16];
      v16h a = join16(*(const v8h*)ap, *(const v8h*)(ap + 8));
      const half_t* bpq = WtQ + ncol * DM + kb + hf * 16;
      const half_t* bpk = WtK + ncol * DM + kb + hf * 16;
      const half_t* bpv = WtV + ncol * DM + kb + hf * 16;
      v16h b_q = join16(*(const v8h*)bpq, *(const v8h*)(bpq + 8));
      v16h b_k = join16(*(const v8h*)bpk, *(const v8h*)(bpk + 8));
      v16h b_v = join16(*(const v8h*)bpv, *(const v8h*)(bpv + 8));
      cq = __builtin_amdgcn_wmma_f32_16x16x32_f16(false, a, false, b_q, (short)0, cq, false, false);
      ck = __builtin_amdgcn_wmma_f32_16x16x32_f16(false, a, false, b_k, (short)0, ck, false, false);
      cv = __builtin_amdgcn_wmma_f32_16x16x32_f16(false, a, false, b_v, (short)0, cv, false, false);
    }
    // ---- RoPE in C-layout (pairwise via DPP swap), stash f16 to LDS ----
    {
      const int n  = lr;
      const int pj = n >> 1;
      const float sgn = (n & 1) ? 1.0f : -1.0f; // even: c*e - s*o ; odd: s*e + c*o
#pragma unroll
      for (int j = 0; j < 8; ++j) {
        const int m = wv * 16 + hf * 8 + j;
        const float sv  = sinS[m * 8 + pj];
        const float cvv = cosS[m * 8 + pj];
        float qv = cq[j], kv = ck[j];
        float qp = swap_pair(qv);
        float kp = swap_pair(kv);
        qh16[m * QSTR + n] = (half_t)(cvv * qv + sgn * sv * qp);
        kh16[m * QSTR + n] = (half_t)(cvv * kv + sgn * sv * kp);
        vT16[n * TSTR + m] = (half_t)cv[j];    // store V transposed: [n][key]
      }
    }
    __syncthreads();

    // ---- scores = q_rot k_rot^T / sqrt(16) : f16 WMMA, K padded 16->32 ----
    {
      const half_t* qp2 = &qh16[arow * QSTR + hf * 16];
      const v16h aq = join16(*(const v8h*)qp2, *(const v8h*)(qp2 + 8));
#pragma unroll
      for (int nt = 0; nt < 8; ++nt) {
        const int key = nt * 16 + lr;
        const half_t* bp = &kh16[key * QSTR + hf * 16];   // lanes>=16 hit the 0 pad
        v16h bk16 = join16(*(const v8h*)bp, *(const v8h*)(bp + 8));
        v8f cs;
#pragma unroll
        for (int i = 0; i < 8; ++i) cs[i] = 0.0f;
        cs = __builtin_amdgcn_wmma_f32_16x16x32_f16(false, aq, false, bk16, (short)0, cs, false, false);
#pragma unroll
        for (int j = 0; j < 8; ++j)
          scB[(wv * 16 + hf * 8 + j) * TSTR + nt * 16 + lr] = (half_t)(cs[j] * 0.25f);
      }
    }
    __syncthreads();

    // ---- softmax over keys: 2 threads per row, pair-combined via DPP ----
    {
      const int row = tid >> 1;
      const int k0  = (tid & 1) * 64;
      half_t* srow = &scB[row * TSTR];
      float mx = -3.4e38f;
      for (int k2 = k0; k2 < k0 + 64; ++k2) mx = fmaxf(mx, (float)srow[k2]);
      mx = fmaxf(mx, swap_pair(mx));
      float s = 0.0f;
      for (int k2 = k0; k2 < k0 + 64; ++k2) {
        float e = __expf((float)srow[k2] - mx);   // e in (0,1], safe in f16
        srow[k2] = (half_t)e;
        s += e;
      }
      s += swap_pair(s);
      float inv = 1.0f / s;
      for (int k2 = k0; k2 < k0 + 64; ++k2)
        srow[k2] = (half_t)((float)srow[k2] * inv);
    }
    __syncthreads();

    // ---- out_head = attn @ v : f16 WMMA, K=128 in 4 steps + pooled colsum ----
    v8f co;
#pragma unroll
    for (int i = 0; i < 8; ++i) co[i] = 0.0f;
#pragma unroll
    for (int kk = 0; kk < 4; ++kk) {
      const int kb = kk * 32;
      const half_t* ap = &scB[arow * TSTR + kb + hf * 16];
      v16h a = join16(*(const v8h*)ap, *(const v8h*)(ap + 8));
      const half_t* bp = &vT16[lr * TSTR + kb + hf * 16]; // B[k][n] = vT16[n][k]
      v16h b_v = join16(*(const v8h*)bp, *(const v8h*)(bp + 8));
      co = __builtin_amdgcn_wmma_f32_16x16x32_f16(false, a, false, b_v, (short)0, co, false, false);
    }
    float ps = 0.0f;
#pragma unroll
    for (int j = 0; j < 8; ++j) ps += co[j];      // sum over this lane's 8 rows
    ps += __shfl_xor(ps, 16, 32);                 // combine the two half-waves
    if (lane < 16) atomicAdd(&poolS[h * DH + lr], ps * (1.0f / SL));
    __syncthreads();  // also guards q/k/v/scB reuse next head
  }

  // ---------------- decoder: 128 -> 64 -> 32 -> 1 ----------------
  if (tid < DHALF) {
    float a = db1[tid];
    for (int k2 = 0; k2 < DM; ++k2) a += poolS[k2] * dW1[k2 * DHALF + tid];
    h1S[tid] = tanhshrink(a);
  }
  __syncthreads();
  if (tid < 32) {
    float a = db2[tid];
    for (int k2 = 0; k2 < DHALF; ++k2) a += h1S[k2] * dW2[k2 * 32 + tid];
    h2S[tid] = tanhshrink(a);
  }
  __syncthreads();
  if (tid == 0) {
    float a = db3[0];
    for (int k2 = 0; k2 < 32; ++k2) a += h2S[k2] * dW3[k2];
    out[b] = a;
  }
}

extern "C" void kernel_launch(void* const* d_in, const int* in_sizes, int n_in,
                              void* d_out, int out_size, void* d_ws, size_t ws_size,
                              hipStream_t stream) {
  const float* x      = (const float*)d_in[0];
  const float* y      = (const float*)d_in[1];
  const float* coords = (const float*)d_in[2];
  const float* W1x = (const float*)d_in[3];
  const float* b1x = (const float*)d_in[4];
  const float* W2x = (const float*)d_in[5];
  const float* b2x = (const float*)d_in[6];
  const float* W1y = (const float*)d_in[7];
  const float* b1y = (const float*)d_in[8];
  const float* W2y = (const float*)d_in[9];
  const float* b2y = (const float*)d_in[10];
  const float* learnY = (const float*)d_in[11];
  const float* Wq = (const float*)d_in[12];
  const float* bq = (const float*)d_in[13];
  const float* Wk = (const float*)d_in[14];
  const float* bk = (const float*)d_in[15];
  const float* Wv = (const float*)d_in[16];
  const float* bv = (const float*)d_in[17];
  const float* dW1 = (const float*)d_in[18];
  const float* db1 = (const float*)d_in[19];
  const float* dW2 = (const float*)d_in[20];
  const float* db2 = (const float*)d_in[21];
  const float* dW3 = (const float*)d_in[22];
  const float* db3 = (const float*)d_in[23];

  half_t* wt = (half_t*)d_ws;  // 61440 halfs = 120 KB scratch

  geo_wprep<<<(WS_TOT + 255) / 256, 256, 0, stream>>>(Wq, Wk, Wv, W1x, W1y, W2x, W2y, wt);
  geo_main<<<BS, 256, 0, stream>>>(x, y, coords, b1x, b2x, b1y, b2y, learnY,
                                   wt + WS_WQ, wt + WS_WK, wt + WS_WV,
                                   wt + WS_W1X, wt + WS_W1Y, wt + WS_W2X, wt + WS_W2Y,
                                   bq, bk, bv, dW1, db1, dW2, db2, dW3, db3,
                                   (float*)d_out);
  (void)in_sizes; (void)n_in; (void)out_size; (void)ws_size;
}
